// MolConv_64037962383975
// MI455X (gfx1250) — matrix operand order repaired and checked by reading, not compiled, not run on hardware
//
#include <hip/hip_runtime.h>
#include <hip/hip_bf16.h>

typedef __attribute__((ext_vector_type(16))) __bf16 v16bf;
typedef __attribute__((ext_vector_type(8)))  float  v8f;

#define F_IN   64
#define KF     256   // K * F_OUT
#define F_OUT  64
#define BN_EPS 1e-5f

__device__ __forceinline__ unsigned int f2bf(float f) {
    union { float f; unsigned int u; } c; c.f = f;
    unsigned int u = c.u;
    u += 0x7FFFu + ((u >> 16) & 1u);   // round-to-nearest-even
    return (u >> 16) & 0xFFFFu;
}
__device__ __forceinline__ float elu(float x) {
    return x > 0.0f ? x : (__expf(x) - 1.0f);
}

// ---------------- Kernel 1: BN column stats (sum, sumsq) ----------------
__global__ void bn_stats_kernel(const float* __restrict__ x,
                                float* __restrict__ stats, int N) {
    __shared__ float s_sum[F_IN];
    __shared__ float s_sq[F_IN];
    int tid = threadIdx.x;
    int col = tid & 63;
    int rg  = tid >> 6;                 // 4 row groups per block
    if (tid < F_IN) { s_sum[tid] = 0.0f; s_sq[tid] = 0.0f; }
    __syncthreads();
    float acc = 0.0f, acc2 = 0.0f;
    for (int r = blockIdx.x * 4 + rg; r < N; r += gridDim.x * 4) {
        float v = x[(size_t)r * F_IN + col];
        acc += v; acc2 += v * v;
    }
    atomicAdd(&s_sum[col], acc);        // ds_add_f32
    atomicAdd(&s_sq[col],  acc2);
    __syncthreads();
    if (tid < F_IN) {
        atomicAdd(&stats[tid],        s_sum[tid]);
        atomicAdd(&stats[F_IN + tid], s_sq[tid]);
    }
}

// ---- Kernel 2: finalize BN scale/shift + repack W into WMMA B-fragments ----
// B-fragment layout used here for v_wmma_f32_16x16x32_bf16 (wave32; contiguous K, A uses the same order):
//   V0..V7; lanes 0-15 hold K=0..15 (V0 = K0,K1 packed), lanes 16-31 hold K=16..31.
//   wfrag[((ktile*16 + ntile)*32 + lane)*8 + v] packs bf16(W[k][n]) | bf16(W[k+1][n])<<16
__global__ void prep_kernel(const float* __restrict__ stats,
                            const float* __restrict__ gamma,
                            const float* __restrict__ beta,
                            const float* __restrict__ W,
                            float* __restrict__ scale_shift,
                            unsigned int* __restrict__ wfrag, int N) {
    if (blockIdx.x == 32) {
        int t = threadIdx.x;
        if (t < F_IN) {
            float invN = 1.0f / (float)N;
            float mean = stats[t] * invN;
            float var  = stats[F_IN + t] * invN - mean * mean;   // biased var
            float inv  = rsqrtf(var + BN_EPS);
            float sc   = gamma[t] * inv;
            scale_shift[t]        = sc;
            scale_shift[F_IN + t] = beta[t] - mean * sc;
        }
        return;
    }
    int idx   = blockIdx.x * 256 + threadIdx.x;   // 0..8191
    int v     = idx & 7;
    int lane  = (idx >> 3) & 31;
    int ntile = (idx >> 8) & 15;
    int ktile = (idx >> 12) & 1;
    int kloc  = ((lane >> 4) << 4) + v * 2;       // 0..30 within 32-K chunk
    int k     = ktile * 32 + kloc;
    int n     = ntile * 16 + (lane & 15);
    unsigned int lo = f2bf(W[(size_t)k * KF + n]);
    unsigned int hi = f2bf(W[(size_t)(k + 1) * KF + n]);
    wfrag[idx] = lo | (hi << 16);
}

// ---------------- Kernel 3: fused BN+ELU + WMMA GEMM -> bf16 x[N,256] ----------------
__global__ void gemm_kernel(const float* __restrict__ atoms,
                            const float* __restrict__ scale_shift,
                            const float* __restrict__ bvec,
                            const unsigned int* __restrict__ wfrag,
                            unsigned int* __restrict__ xout, int N) {
    __shared__ unsigned int   s_a[512];    // 16 rows x 64 bf16 (packed pairs)
    __shared__ unsigned short s_o[4096];   // 16 rows x 256 bf16 output tile
    int tid  = threadIdx.x;
    int row0 = blockIdx.x * 16;
    if (row0 + 16 > N) row0 = (N >= 16) ? (N - 16) : 0;

    // Stage A-tile: 1024 floats contiguous (16 rows x 64 cols), BN+ELU, -> bf16 LDS
    {
        const float4* src = (const float4*)(atoms + (size_t)row0 * F_IN);
        float4 f = src[tid];
        int c0 = (tid * 4) & 63;
        float e0 = elu(f.x * scale_shift[c0 + 0] + scale_shift[64 + c0 + 0]);
        float e1 = elu(f.y * scale_shift[c0 + 1] + scale_shift[64 + c0 + 1]);
        float e2 = elu(f.z * scale_shift[c0 + 2] + scale_shift[64 + c0 + 2]);
        float e3 = elu(f.w * scale_shift[c0 + 3] + scale_shift[64 + c0 + 3]);
        s_a[tid * 2 + 0] = f2bf(e0) | (f2bf(e1) << 16);
        s_a[tid * 2 + 1] = f2bf(e2) | (f2bf(e3) << 16);
    }
    __syncthreads();

    int lane = tid & 31, wid = tid >> 5;
    int m = lane & 15, hig = lane >> 4;

    // A-fragments (16-bit A 16x32, contiguous K to match wfrag): pair index p = v + hig*8
    union AF { v16bf v; unsigned int u[8]; } a0, a1;
#pragma unroll
    for (int v = 0; v < 8; ++v) {
        int p = v + hig * 8;
        a0.u[v] = s_a[m * 32 + p];        // K = 0..31
        a1.u[v] = s_a[m * 32 + 16 + p];   // K = 32..63
    }

#pragma unroll
    for (int t = 0; t < 2; ++t) {
        int ntile = wid + t * 8;          // 8 waves x 2 = 16 column tiles
        union BF { v16bf v; uint4 q[2]; } b0, b1;
        const uint4* wp0 = (const uint4*)(wfrag + ((size_t)(0 * 16 + ntile) * 32 + lane) * 8);
        const uint4* wp1 = (const uint4*)(wfrag + ((size_t)(1 * 16 + ntile) * 32 + lane) * 8);
        b0.q[0] = wp0[0]; b0.q[1] = wp0[1];
        b1.q[0] = wp1[0]; b1.q[1] = wp1[1];

        v8f c = {};
        c = __builtin_amdgcn_wmma_f32_16x16x32_bf16(false, a0.v, false, b0.v,
                                                    (short)0, c, false, false);
        c = __builtin_amdgcn_wmma_f32_16x16x32_bf16(false, a1.v, false, b1.v,
                                                    (short)0, c, false, false);

        int n    = ntile * 16 + m;
        float bb = bvec[n];
        // C layout: VGPR i -> (M=i, lanes 0-15) / (M=i+8, lanes 16-31), N=lane&15.
        // Repack through LDS (ds_store_b16, 32-bit addressing) instead of 2-byte
        // global stores with 64-bit address math.
#pragma unroll
        for (int i = 0; i < 8; ++i) {
            int M = i + hig * 8;
            s_o[M * KF + n] = (unsigned short)f2bf(c[i] + bb);
        }
    }
    __syncthreads();

    // Coalesced writeback: 16 rows x 256 bf16 = 2048 u32 contiguous at row0*128
    {
        const uint4* so4 = (const uint4*)s_o;
        uint4* dst = (uint4*)(xout + (size_t)row0 * (KF / 2));
        dst[tid * 2 + 0] = so4[tid * 2 + 0];   // global_store_b128
        dst[tid * 2 + 1] = so4[tid * 2 + 1];
    }
}

// ---------------- Kernel 4: SpMM gather + atomic scatter ----------------
// stacked[c] with c in [0, K*N): n = c % N, k = c / N -> x[n, k*64 + f]
#define SPMM_LOOKAHEAD 2048
__global__ void spmm_kernel(const unsigned short* __restrict__ xbf,
                            const float* __restrict__ vals,
                            const int* __restrict__ rows,
                            const int* __restrict__ cols,
                            float* __restrict__ out, int E, int N) {
    unsigned long long t = (unsigned long long)blockIdx.x * 256ull + threadIdx.x;
    int e   = (int)(t >> 4);
    int sub = (int)(t & 15);
    if (e >= E) return;

    // Look-ahead prefetch of a future edge's gather row (global_prefetch_b8).
    // x (51 MB bf16) is L2-resident; this pulls the 64B half-lines toward the WGP.
    if ((sub & 7) == 0) {
        int ela = e + SPMM_LOOKAHEAD;
        if (ela < E) {
            unsigned int cla = (unsigned int)cols[ela];
            unsigned int kla = cla / (unsigned int)N;
            unsigned int nla = cla - kla * (unsigned int)N;
            __builtin_prefetch(xbf + (size_t)nla * KF + kla * F_OUT + sub * 4, 0, 1);
        }
    }

    int r = rows[e];
    unsigned int c = (unsigned int)cols[e];
    float v = vals[e];
    unsigned int k = c / (unsigned int)N;
    unsigned int n = c - k * (unsigned int)N;

    const uint2* src = (const uint2*)(xbf + (size_t)n * KF + k * F_OUT + sub * 4);
    uint2 p = *src;                        // 4 bf16 = global_load_b64

    float f0 = __uint_as_float(p.x << 16);
    float f1 = __uint_as_float(p.x & 0xFFFF0000u);
    float f2 = __uint_as_float(p.y << 16);
    float f3 = __uint_as_float(p.y & 0xFFFF0000u);

    float* o = out + (size_t)r * F_OUT + sub * 4;
    unsafeAtomicAdd(o + 0, v * f0);        // global_atomic_add_f32
    unsafeAtomicAdd(o + 1, v * f1);
    unsafeAtomicAdd(o + 2, v * f2);
    unsafeAtomicAdd(o + 3, v * f3);
}

// ---------------- Host launch ----------------
extern "C" void kernel_launch(void* const* d_in, const int* in_sizes, int n_in,
                              void* d_out, int out_size, void* d_ws, size_t ws_size,
                              hipStream_t stream) {
    const float* atoms    = (const float*)d_in[0];
    const float* gamma    = (const float*)d_in[1];
    const float* beta     = (const float*)d_in[2];
    const float* W        = (const float*)d_in[3];
    const float* bvec     = (const float*)d_in[4];
    const float* adj_vals = (const float*)d_in[5];
    const int*   adj_rows = (const int*)d_in[6];
    const int*   adj_cols = (const int*)d_in[7];

    int N = in_sizes[0] / F_IN;
    int E = in_sizes[5];
    float* out = (float*)d_out;

    // Workspace layout:
    //   [0,512)            : stats (sum[64], sumsq[64])
    //   [512,1024)         : scale[64], shift[64]
    //   [1024,1024+32K)    : W fragments (8192 u32)
    //   [33792, +N*256*2)  : bf16 activations x[N,256]  (~51.2 MB, L2-resident)
    char* ws = (char*)d_ws;
    float*        stats       = (float*)ws;
    float*        scale_shift = (float*)(ws + 512);
    unsigned int* wfrag       = (unsigned int*)(ws + 1024);
    unsigned int* xbf_u32     = (unsigned int*)(ws + 1024 + 32768);

    hipMemsetAsync(stats, 0, 512, stream);
    hipMemsetAsync(d_out, 0, (size_t)out_size * sizeof(float), stream);

    bn_stats_kernel<<<256, 256, 0, stream>>>(atoms, stats, N);
    prep_kernel<<<33, 256, 0, stream>>>(stats, gamma, beta, W, scale_shift, wfrag, N);
    gemm_kernel<<<(N + 15) / 16, 256, 0, stream>>>(atoms, scale_shift, bvec, wfrag,
                                                   xbf_u32, N);

    int spmm_blocks = (int)(((long long)E * 16 + 255) / 256);
    spmm_kernel<<<spmm_blocks, 256, 0, stream>>>((const unsigned short*)xbf_u32,
                                                 adj_vals, adj_rows, adj_cols, out, E, N);
}
